// DSQN_61864708931767
// MI455X (gfx1250) — matrix operand order repaired and checked by reading, not compiled, not run on hardware
//
#include <hip/hip_runtime.h>
#include <cstdint>

// CDNA5 / gfx1250, wave32. D = A*B + C via v_wmma_f32_16x16x32_f16.
typedef __attribute__((ext_vector_type(16))) _Float16 v16h;
typedef __attribute__((ext_vector_type(8)))  float    v8f;

#define TSTEPS 1024
#define FDIM   128
#define HDIM   128
#define ADIM   16
#define ROWS   16      // batch rows per workgroup (one WMMA M-tile)

union AFragU { v16h v; uint32_t u[8]; _Float16 h[16]; };

// A-matrix fragment (16x32, f16) from row-major LDS tile [16][128].
// Layout (ISA 7.12.2): lanes 0-15 row M=lane, VGPR i holds K pair
//   (i<4 ? 2i : 16+2(i-4)) (+8 for lanes 16-31, M=lane-16).
__device__ __forceinline__ v16h load_a_frag(const _Float16* s, int lane, int kc) {
  const int r  = lane & 15;
  const int hh = (lane >> 4) * 8;
  const _Float16* row = s + r * FDIM + kc * 32;
  AFragU f;
#pragma unroll
  for (int i = 0; i < 8; ++i) {
    int k = (i < 4 ? 2 * i : 16 + 2 * (i - 4)) + hh;
    f.u[i] = *(const uint32_t*)(row + k);                 // two contiguous f16
  }
  return f.v;
}

// B-matrix fragment (32x16, f16) for B[k][n] = W[n][k] (W row-major [*][128]).
// Layout: VGPR i: lanes 0-15 (N=lane) K pair (i<4 ? 2i : 16+2(i-4)); lanes 16-31 K+=8.
__device__ __forceinline__ v16h make_b_frag(const float* W, int n, int kc, int lane) {
  const int kh = (lane >> 4) << 3;
  const float* row = W + (size_t)n * HDIM + kc * 32;      // HDIM == FDIM == 128
  AFragU f;
#pragma unroll
  for (int i = 0; i < 8; ++i) {
    int k = (i < 4 ? 2 * i : 16 + 2 * (i - 4)) + kh;
    f.h[2 * i]     = (_Float16)row[k];
    f.h[2 * i + 1] = (_Float16)row[k + 1];
  }
  return f.v;
}

__global__ __launch_bounds__(256)
void dsqn_scan_kernel(const float* __restrict__ state,   // (B,T,F)
                      const float* __restrict__ hs,      // (B,1,2,H)
                      const float* __restrict__ W1, const float* __restrict__ b1,
                      const float* __restrict__ beta1,
                      const float* __restrict__ W2, const float* __restrict__ b2,
                      const float* __restrict__ beta2,
                      const float* __restrict__ W3, const float* __restrict__ b3,
                      float* __restrict__ out) {         // (B,T,A) f32
  __shared__ _Float16 sState[2][ROWS * FDIM];            // double buffered
  __shared__ _Float16 sSpk1[ROWS * HDIM];
  __shared__ _Float16 sSpk2[ROWS * HDIM];

  const int tid  = threadIdx.x;
  const int lane = tid & 31;
  const int w    = tid >> 5;            // wave 0..7 -> hidden column tile
  const int b0   = blockIdx.x * ROWS;   // batch tile base

  const int nl   = lane & 15;           // N within tile (B/C/D frag column)
  const int hcol = w * 16 + nl;         // this wave's global hidden column
  const int rhi  = (lane >> 4) * 8;     // C-frag: M = vgpr + rhi

  // ---- loop-invariant fragments & per-lane scalars (built once) ----
  v16h fW1[4], fW2[4], fW3[4];
#pragma unroll
  for (int kc = 0; kc < 4; ++kc) {
    fW1[kc] = make_b_frag(W1, hcol, kc, lane);
    fW2[kc] = make_b_frag(W2, hcol, kc, lane);
    fW3[kc] = make_b_frag(W3, nl,   kc, lane);   // used by wave 0 only
  }
  const float bt1 = fminf(fmaxf(beta1[hcol], 0.f), 1.f);
  const float bt2 = fminf(fmaxf(beta2[hcol], 0.f), 1.f);
  const float bb1 = b1[hcol], bb2 = b2[hcol], bb3 = b3[nl];

  // ---- membrane state resident in C-fragment registers ----
  v8f mem1, mem2;
#pragma unroll
  for (int v = 0; v < 8; ++v) {
    const size_t b = (size_t)(b0 + v + rhi);
    mem1[v] = hs[b * 2 * HDIM + hcol];
    mem2[v] = hs[b * 2 * HDIM + HDIM + hcol];
  }

  // state-tile loader assignment: 256 thr * 8 elems = 16x128
  const int sr = tid >> 4;              // row 0..15
  const int sf = (tid & 15) * 8;        // col base 0..120
  const float* gbase = state + ((size_t)(b0 + sr) * TSTEPS) * FDIM + sf;

  // ---- prologue: stage state tile for t=0 ----
  {
    float4 x0 = *(const float4*)gbase;
    float4 x1 = *(const float4*)(gbase + 4);
    _Float16* d = sState[0] + sr * FDIM + sf;
    d[0]=(_Float16)x0.x; d[1]=(_Float16)x0.y; d[2]=(_Float16)x0.z; d[3]=(_Float16)x0.w;
    d[4]=(_Float16)x1.x; d[5]=(_Float16)x1.y; d[6]=(_Float16)x1.z; d[7]=(_Float16)x1.w;
  }
  __syncthreads();

  for (int t = 0; t < TSTEPS; ++t) {
    const int p = t & 1;

    // issue global loads for step t+1 (land far from use; prefetch reaches t+2)
    float4 nx0, nx1;
    if (t + 1 < TSTEPS) {
      const float* g = gbase + (size_t)(t + 1) * FDIM;
      nx0 = *(const float4*)g;
      nx1 = *(const float4*)(g + 4);
      __builtin_prefetch(g + FDIM, 0, 1);              // global_prefetch_b8 (t+2)
    }

    // cur1 = state @ W1^T + b1 : issue all A-frag LDS loads, then WMMA chain
    v16h a0 = load_a_frag(sState[p], lane, 0);
    v16h a1 = load_a_frag(sState[p], lane, 1);
    v16h a2 = load_a_frag(sState[p], lane, 2);
    v16h a3 = load_a_frag(sState[p], lane, 3);
    v8f c1 = { bb1, bb1, bb1, bb1, bb1, bb1, bb1, bb1 };
    c1 = __builtin_amdgcn_wmma_f32_16x16x32_f16(false, a0, false, fW1[0], (short)0, c1, false, false);
    c1 = __builtin_amdgcn_wmma_f32_16x16x32_f16(false, a1, false, fW1[1], (short)0, c1, false, false);
    c1 = __builtin_amdgcn_wmma_f32_16x16x32_f16(false, a2, false, fW1[2], (short)0, c1, false, false);
    c1 = __builtin_amdgcn_wmma_f32_16x16x32_f16(false, a3, false, fW1[3], (short)0, c1, false, false);

    // LIF layer 1: reset uses OLD mem; THR = 1
#pragma unroll
    for (int v = 0; v < 8; ++v) {
      float m     = mem1[v];
      float reset = (m > 1.0f) ? 1.0f : 0.0f;
      float mn    = __builtin_fmaf(bt1, m, c1[v]) - reset;
      mem1[v]     = mn;
      sSpk1[(v + rhi) * HDIM + hcol] = (_Float16)((mn > 1.0f) ? 1.0f : 0.0f);
    }

    // stage next step's state tile into the alternate buffer
    if (t + 1 < TSTEPS) {
      _Float16* d = sState[p ^ 1] + sr * FDIM + sf;
      d[0]=(_Float16)nx0.x; d[1]=(_Float16)nx0.y; d[2]=(_Float16)nx0.z; d[3]=(_Float16)nx0.w;
      d[4]=(_Float16)nx1.x; d[5]=(_Float16)nx1.y; d[6]=(_Float16)nx1.z; d[7]=(_Float16)nx1.w;
    }
    __syncthreads();                                   // barrier A

    // cur2 = spk1 @ W2^T + b2
    a0 = load_a_frag(sSpk1, lane, 0);
    a1 = load_a_frag(sSpk1, lane, 1);
    a2 = load_a_frag(sSpk1, lane, 2);
    a3 = load_a_frag(sSpk1, lane, 3);
    v8f c2 = { bb2, bb2, bb2, bb2, bb2, bb2, bb2, bb2 };
    c2 = __builtin_amdgcn_wmma_f32_16x16x32_f16(false, a0, false, fW2[0], (short)0, c2, false, false);
    c2 = __builtin_amdgcn_wmma_f32_16x16x32_f16(false, a1, false, fW2[1], (short)0, c2, false, false);
    c2 = __builtin_amdgcn_wmma_f32_16x16x32_f16(false, a2, false, fW2[2], (short)0, c2, false, false);
    c2 = __builtin_amdgcn_wmma_f32_16x16x32_f16(false, a3, false, fW2[3], (short)0, c2, false, false);

#pragma unroll
    for (int v = 0; v < 8; ++v) {
      float m     = mem2[v];
      float reset = (m > 1.0f) ? 1.0f : 0.0f;
      float mn    = __builtin_fmaf(bt2, m, c2[v]) - reset;
      mem2[v]     = mn;
      sSpk2[(v + rhi) * HDIM + hcol] = (_Float16)((mn > 1.0f) ? 1.0f : 0.0f);
    }
    __syncthreads();                                   // barrier B

    // out = spk2 @ W3^T + b3  (16x16 tile; wave 0, overlaps others' next cur1)
    if (w == 0) {
      v16h q0 = load_a_frag(sSpk2, lane, 0);
      v16h q1 = load_a_frag(sSpk2, lane, 1);
      v16h q2 = load_a_frag(sSpk2, lane, 2);
      v16h q3 = load_a_frag(sSpk2, lane, 3);
      v8f c3 = { bb3, bb3, bb3, bb3, bb3, bb3, bb3, bb3 };
      c3 = __builtin_amdgcn_wmma_f32_16x16x32_f16(false, q0, false, fW3[0], (short)0, c3, false, false);
      c3 = __builtin_amdgcn_wmma_f32_16x16x32_f16(false, q1, false, fW3[1], (short)0, c3, false, false);
      c3 = __builtin_amdgcn_wmma_f32_16x16x32_f16(false, q2, false, fW3[2], (short)0, c3, false, false);
      c3 = __builtin_amdgcn_wmma_f32_16x16x32_f16(false, q3, false, fW3[3], (short)0, c3, false, false);
#pragma unroll
      for (int v = 0; v < 8; ++v) {
        out[((size_t)(b0 + v + rhi) * TSTEPS + t) * ADIM + nl] = c3[v];
      }
    }
  }
}

extern "C" void kernel_launch(void* const* d_in, const int* in_sizes, int n_in,
                              void* d_out, int out_size, void* d_ws, size_t ws_size,
                              hipStream_t stream) {
  const float* state = (const float*)d_in[0];
  const float* hs    = (const float*)d_in[1];
  const float* W1    = (const float*)d_in[2];
  const float* b1    = (const float*)d_in[3];
  const float* beta1 = (const float*)d_in[4];
  const float* W2    = (const float*)d_in[5];
  const float* b2    = (const float*)d_in[6];
  const float* beta2 = (const float*)d_in[7];
  const float* W3    = (const float*)d_in[8];
  const float* b3    = (const float*)d_in[9];
  float* out = (float*)d_out;

  const int B = in_sizes[1] / (2 * HDIM);   // hidden_states = (B,1,2,H)
  dim3 grid(B / ROWS), block(256);
  dsqn_scan_kernel<<<grid, block, 0, stream>>>(state, hs, W1, b1, beta1,
                                               W2, b2, beta2, W3, b3, out);
}
